// Qwen3VLVisionTransformer_19138374271042
// MI455X (gfx1250) — compile-verified, matrix operand order checked by reading, not run on hardware
//
#include <hip/hip_runtime.h>
#include <math.h>

typedef __attribute__((ext_vector_type(16))) __bf16 v16bf;
typedef __attribute__((ext_vector_type(8)))  float  v8f;
typedef __attribute__((ext_vector_type(4)))  int    v4i;

#define T_TOK   3072
#define D_MODEL 1152
#define D_FF    4304
#define NHEAD   16
#define HDIM    72
#define HPAD    96
#define NEG_INF_F (-1.0e30f)

#ifndef __has_builtin
#define __has_builtin(x) 0
#endif
#if __has_builtin(__builtin_amdgcn_global_load_async_to_lds_b128) && \
    __has_builtin(__builtin_amdgcn_s_wait_asynccnt)
#define ASYNC_COPY 1
#else
#define ASYNC_COPY 0
#endif

// ---------- helpers ----------
static __device__ __forceinline__ unsigned short f2bf(float f) {
  unsigned u = __float_as_uint(f);
  u += 0x7FFFu + ((u >> 16) & 1u);          // round-to-nearest-even
  return (unsigned short)(u >> 16);
}

#if ASYNC_COPY
typedef __attribute__((address_space(1))) v4i gv4i;   // global int4
typedef __attribute__((address_space(3))) v4i lv4i;   // LDS int4
// one 16B chunk per active lane: LDS[l] = GLOBAL[g]; tracked by ASYNCcnt
static __device__ __forceinline__ void async_b128(const void* g, void* l) {
  __builtin_amdgcn_global_load_async_to_lds_b128((gv4i*)g, (lv4i*)l, 0, 0);
}
#endif

// Load a 16x32 bf16 WMMA A/B fragment from an LDS tile.
// ISA layout: lanes 0-15 hold row M=lane, K={0..7,16..23}; lanes 16-31 hold
// row M=lane-16, K={8..15,24..31}. Each half is two contiguous 16B chunks.
static __device__ __forceinline__ v16bf ldfrag(const unsigned short* t, int row,
                                               int stride, int k0, int lane) {
  union { v16bf v; uint4 q[2]; } u;
  int kh = (lane >> 4) & 1;
  const unsigned short* p = t + row * stride + k0 + kh * 8;
  u.q[0] = *(const uint4*)p;
  u.q[1] = *(const uint4*)(p + 16);
  return u.v;
}

static __device__ __forceinline__ v8f wmma_bf16(v16bf a, v16bf b, v8f c) {
  return __builtin_amdgcn_wmma_f32_16x16x32_bf16(false, a, false, b,
                                                 (short)0, c, false, false);
}

// ---------- fp32 -> bf16 conversion ----------
__global__ void cvt_bf16_kernel(const float* __restrict__ src,
                                unsigned short* __restrict__ dst, long n) {
  long i = (long)blockIdx.x * blockDim.x + threadIdx.x;
  long stride = (long)gridDim.x * blockDim.x;
  for (; i < n; i += stride) dst[i] = f2bf(src[i]);
}

// ---------- RMS norm (row per block), bf16 output ----------
__global__ __launch_bounds__(256) void rmsnorm_kernel(
    const float* __restrict__ x, const float* __restrict__ scale,
    unsigned short* __restrict__ out) {
  __shared__ float red[8];
  const int row = blockIdx.x, tid = threadIdx.x;
  const float* xr = x + (long)row * D_MODEL;
  float s = 0.0f;
  for (int i = tid; i < D_MODEL; i += 256) { float v = xr[i]; s += v * v; }
#pragma unroll
  for (int m = 16; m >= 1; m >>= 1) s += __shfl_xor(s, m, 32);
  if ((tid & 31) == 0) red[tid >> 5] = s;
  __syncthreads();
  float tot = 0.0f;
#pragma unroll
  for (int wv = 0; wv < 8; ++wv) tot += red[wv];
  float inv = rsqrtf(tot / (float)D_MODEL + 1e-6f);
  for (int i = tid; i < D_MODEL; i += 256)
    out[(long)row * D_MODEL + i] = f2bf(xr[i] * inv * scale[i]);
}

// ---------- generic bf16 WMMA GEMM: C = A(MxK) @ B(KxN) + bias ----------
// Double-buffered LDS; A tile via async global->LDS DMA (when available),
// B tile via register double-buffer with transpose-on-deposit.
// EPI: 0 = bias -> f32 out ; 1 = bias+gelu -> bf16 out ; 2 = bias+res -> f32 out
template <int EPI>
__global__ __launch_bounds__(256) void gemm_bf16_wmma(
    const unsigned short* __restrict__ A, const unsigned short* __restrict__ B,
    const float* __restrict__ bias, const float* __restrict__ res,
    float* __restrict__ outF, unsigned short* __restrict__ outB,
    int M, int N, int K) {
  __shared__ __align__(16) unsigned short As[2 * 128 * 32];   // [buf][m][k]
  __shared__ __align__(16) unsigned short Bs[2 * 128 * 32];   // [buf][n][k] (transposed)
  const int tid = threadIdx.x;
  const int lane = tid & 31, wid = tid >> 5;
  const int wr = wid & 3, wc = wid >> 2;          // wave tile: 32 rows x 64 cols
  const int m0 = blockIdx.y * 128, n0 = blockIdx.x * 128;
  const int lane15 = lane & 15, kh = lane >> 4;
  const int KT = (K + 31) / 32;

  v8f acc[2][4] = {};
  uint4 bReg[2];
#if !ASYNC_COPY
  uint4 aReg[2];
#endif

  // ---- prologue: stage k-tile 0 ----
#pragma unroll
  for (int i = 0; i < 2; ++i) {              // B tile 0 -> regs
    int c = tid + 256 * i;
    int krow = c >> 4, nc = (c & 15) * 8;
    int gn = n0 + nc;
    bReg[i] = make_uint4(0u, 0u, 0u, 0u);
    if (gn < N) bReg[i] = *(const uint4*)(B + (long)krow * N + gn);
  }
#pragma unroll
  for (int i = 0; i < 2; ++i) {              // A tile 0 -> LDS buf0 (or regs)
    int c = tid + 256 * i;
    int row = c >> 2, kc = (c & 3) * 8;
    int gk = kc < K - 8 ? kc : K - 8;        // clamp: OOB A cols hit zeroed B rows
#if ASYNC_COPY
    async_b128(A + (long)(m0 + row) * K + gk, As + c * 8);
#else
    aReg[i] = *(const uint4*)(A + (long)(m0 + row) * K + gk);
#endif
  }

  for (int kt = 0; kt < KT; ++kt) {
    const int cur = kt & 1;
    unsigned short* Ac = As + cur * (128 * 32);
    unsigned short* Bc = Bs + cur * (128 * 32);
    unsigned short* An = As + (cur ^ 1) * (128 * 32);

    // deposit B regs into Bc with transpose
#pragma unroll
    for (int i = 0; i < 2; ++i) {
      int c = tid + 256 * i;
      int krow = c >> 4, nc = (c & 15) * 8;
      union { uint4 q; unsigned short s[8]; } u;
      u.q = bReg[i];
#pragma unroll
      for (int e = 0; e < 8; ++e) Bc[(nc + e) * 32 + krow] = u.s[e];
    }
#if !ASYNC_COPY
#pragma unroll
    for (int i = 0; i < 2; ++i) ((uint4*)Ac)[tid + 256 * i] = aReg[i];
#endif

#if ASYNC_COPY
    if (kt + 1 < KT) {
      const int gk0 = (kt + 1) * 32;
#pragma unroll
      for (int i = 0; i < 2; ++i) {          // async-stream next A tile
        int c = tid + 256 * i;
        int row = c >> 2, kc = (c & 3) * 8;
        int gk = gk0 + kc;
        if (gk > K - 8) gk = K - 8;          // clamped (see above)
        async_b128(A + (long)(m0 + row) * K + gk, An + c * 8);
      }
      __builtin_amdgcn_s_wait_asynccnt(2);   // current tile landed; next in flight
    } else {
      __builtin_amdgcn_s_wait_asynccnt(0);
    }
#endif
    __syncthreads();

    if (kt + 1 < KT) {                       // next B tile -> regs (overlaps WMMA)
      const int gk0 = (kt + 1) * 32;
#pragma unroll
      for (int i = 0; i < 2; ++i) {
        int c = tid + 256 * i;
        int krow = c >> 4, nc = (c & 15) * 8;
        int gk = gk0 + krow, gn = n0 + nc;
        bReg[i] = make_uint4(0u, 0u, 0u, 0u);
        if (gk < K && gn < N) {
          bReg[i] = *(const uint4*)(B + (long)gk * N + gn);
          if (kt + 2 < KT) __builtin_prefetch(B + (long)(gk + 32) * N + gn, 0, 1);
        }
      }
#if !ASYNC_COPY
#pragma unroll
      for (int i = 0; i < 2; ++i) {
        int c = tid + 256 * i;
        int row = c >> 2, kc = (c & 3) * 8;
        int gk = gk0 + kc;
        if (gk > K - 8) gk = K - 8;
        aReg[i] = *(const uint4*)(A + (long)(m0 + row) * K + gk);
      }
#endif
    }

    v16bf bfr[4];
#pragma unroll
    for (int c2 = 0; c2 < 4; ++c2)
      bfr[c2] = ldfrag(Bc, (wc * 4 + c2) * 16 + lane15, 32, 0, lane);
#pragma unroll
    for (int r = 0; r < 2; ++r) {
      v16bf af = ldfrag(Ac, (wr * 2 + r) * 16 + lane15, 32, 0, lane);
#pragma unroll
      for (int c2 = 0; c2 < 4; ++c2)
        acc[r][c2] = wmma_bf16(af, bfr[c2], acc[r][c2]);
    }
    __syncthreads();
  }

  // epilogue: C frag layout = VGPR i -> M=i(+8 for lanes 16-31), N=lane&15
#pragma unroll
  for (int r = 0; r < 2; ++r) {
#pragma unroll
    for (int c2 = 0; c2 < 4; ++c2) {
      int tn = n0 + (wc * 4 + c2) * 16 + lane15;
      if (tn >= N) continue;
      float bv = bias[tn];
      int tm = m0 + (wr * 2 + r) * 16;
#pragma unroll
      for (int i = 0; i < 8; ++i) {
        long row = tm + i + 8 * kh;
        float v = acc[r][c2][i] + bv;
        long idx = row * (long)N + tn;
        if (EPI == 0) {
          outF[idx] = v;
        } else if (EPI == 1) {
          float g = 0.5f * v * (1.0f + erff(v * 0.70710678118f));  // exact gelu
          outB[idx] = f2bf(g);
        } else {
          outF[idx] = res[idx] + v;
        }
      }
    }
  }
}

// ---------- RoPE + pack q/k/v into bf16 attention layouts ----------
// Qb/Kb: [head][T][96] (sm_scale folded into Q), Vt: [head][96][T] (transposed)
__global__ __launch_bounds__(256) void rope_pack_kernel(
    const float* __restrict__ qkv, const float* __restrict__ rope,
    unsigned short* __restrict__ Qb, unsigned short* __restrict__ Kb,
    unsigned short* __restrict__ Vt) {
  const int t = blockIdx.x;
  const int tid = threadIdx.x;
  const float sm = 0.11785113019775793f;  // 1/sqrt(72)
  for (int idx = tid; idx < NHEAD * 36; idx += 256) {
    int n = idx / 36, h = idx - n * 36;
    float ang = rope[t * 36 + h];
    float c = __cosf(ang), s = __sinf(ang);
    const float* qp = qkv + (long)t * (3 * D_MODEL) + n * HDIM;
    float qr = qp[h], qi = qp[h + 36];
    float kr = qp[D_MODEL + h], ki = qp[D_MODEL + h + 36];
    long ob = ((long)n * T_TOK + t) * HPAD;
    Qb[ob + h]      = f2bf((qr * c - qi * s) * sm);
    Qb[ob + h + 36] = f2bf((qr * s + qi * c) * sm);
    Kb[ob + h]      = f2bf(kr * c - ki * s);
    Kb[ob + h + 36] = f2bf(kr * s + ki * c);
  }
  for (int idx = tid; idx < NHEAD * (HPAD - HDIM); idx += 256) {  // zero pad
    int n = idx / (HPAD - HDIM), h = HDIM + idx % (HPAD - HDIM);
    long ob = ((long)n * T_TOK + t) * HPAD;
    Qb[ob + h] = 0; Kb[ob + h] = 0;
  }
  for (int idx = tid; idx < NHEAD * HPAD; idx += 256) {  // V transposed + pad
    int n = idx / HPAD, h = idx - n * HPAD;
    float v = (h < HDIM) ? qkv[(long)t * (3 * D_MODEL) + 2 * D_MODEL + n * HDIM + h]
                         : 0.0f;
    Vt[((long)n * HPAD + h) * T_TOK + t] = f2bf(v);
  }
}

// ---------- flash attention: per (query-block of 128, head) ----------
// 8 waves; wave w owns rows w*16..w*16+15 -> whole key row in one wave.
__global__ __launch_bounds__(256) void attn_kernel(
    const unsigned short* __restrict__ Qb, const unsigned short* __restrict__ Kb,
    const unsigned short* __restrict__ Vt, const int* __restrict__ seg,
    unsigned short* __restrict__ outA) {
  extern __shared__ __align__(16) unsigned char smem[];
  unsigned short* Qs  = (unsigned short*)smem;     // 128 x 96
  unsigned short* KVs = Qs + 128 * HPAD;           // K tile [128][96] / V tile [96][128]
  unsigned short* Ps  = KVs + 128 * HPAD;          // 8 waves x 16 x 128 (bf16 P)
  int* qseg = (int*)(Ps + 8 * 16 * 128);
  int* kseg = qseg + 128;

  const int qb = blockIdx.x, head = blockIdx.y;
  const int tid = threadIdx.x, lane = tid & 31, w = tid >> 5;
  const int lane15 = lane & 15, kh = lane >> 4;

  const unsigned short* qsrc = Qb + ((long)head * T_TOK + qb * 128) * HPAD;
#if ASYNC_COPY
  for (int i = tid; i < 128 * HPAD / 8; i += 256)
    async_b128(qsrc + (long)i * 8, Qs + i * 8);
#else
  for (int i = tid; i < 128 * HPAD / 8; i += 256)
    ((uint4*)Qs)[i] = ((const uint4*)qsrc)[i];
#endif
  if (tid < 128) qseg[tid] = seg[qb * 128 + tid];

  v8f oacc[6] = {};
  float mrun[8], lrun[8];
#pragma unroll
  for (int i = 0; i < 8; ++i) { mrun[i] = -3.0e38f; lrun[i] = 0.0f; }

  unsigned short* Pw = Ps + w * 16 * 128;

  for (int j = 0; j < T_TOK / 128; ++j) {
    __syncthreads();
    const unsigned short* ksrc = Kb + ((long)head * T_TOK + j * 128) * HPAD;
#if ASYNC_COPY
    for (int i = tid; i < 128 * HPAD / 8; i += 256)
      async_b128(ksrc + (long)i * 8, KVs + i * 8);
#else
    for (int i = tid; i < 128 * HPAD / 8; i += 256)
      ((uint4*)KVs)[i] = ((const uint4*)ksrc)[i];
#endif
    if (tid < 128) kseg[tid] = seg[j * 128 + tid];
#if ASYNC_COPY
    __builtin_amdgcn_s_wait_asynccnt(0);   // also covers the Q-tile DMA (iter 0)
#endif
    __syncthreads();

    // S = (Q*sm) @ K^T  (K dim = 96 -> 3 WMMA steps)
    v8f sacc[8] = {};
#pragma unroll
    for (int kk = 0; kk < 3; ++kk) {
      v16bf aq = ldfrag(Qs, w * 16 + lane15, HPAD, kk * 32, lane);
#pragma unroll
      for (int tI = 0; tI < 8; ++tI) {
        v16bf bk = ldfrag(KVs, tI * 16 + lane15, HPAD, kk * 32, lane);
        sacc[tI] = wmma_bf16(aq, bk, sacc[tI]);
      }
    }

    // segment mask + online softmax; write P (bf16) to wave-local LDS
#pragma unroll
    for (int i = 0; i < 8; ++i) {
      int qs = qseg[w * 16 + i + 8 * kh];
      float sv[8];
      float rmax = -3.0e38f;
#pragma unroll
      for (int tI = 0; tI < 8; ++tI) {
        float s = sacc[tI][i];
        if (kseg[tI * 16 + lane15] != qs) s = NEG_INF_F;
        sv[tI] = s;
        rmax = fmaxf(rmax, s);
      }
#pragma unroll
      for (int mm = 1; mm < 16; mm <<= 1)
        rmax = fmaxf(rmax, __shfl_xor(rmax, mm, 32));
      float mnew = fmaxf(mrun[i], rmax);
      float alpha = __expf(mrun[i] - mnew);
      mrun[i] = mnew;
      float rs = 0.0f;
#pragma unroll
      for (int tI = 0; tI < 8; ++tI) {
        float p = __expf(sv[tI] - mnew);
        rs += p;
        Pw[(i + 8 * kh) * 128 + tI * 16 + lane15] = f2bf(p);
      }
#pragma unroll
      for (int mm = 1; mm < 16; mm <<= 1) rs += __shfl_xor(rs, mm, 32);
      lrun[i] = lrun[i] * alpha + rs;
#pragma unroll
      for (int d = 0; d < 6; ++d) oacc[d][i] *= alpha;
    }

    __syncthreads();
    // V tile: Vt[head][dim][T] -> KVs[dim][128], rows contiguous
    const unsigned short* vsrc = Vt + (long)head * HPAD * T_TOK + (long)j * 128;
#if ASYNC_COPY
    for (int i = tid; i < HPAD * 128 / 8; i += 256) {
      int d = i >> 4, c8 = i & 15;
      async_b128(vsrc + (long)d * T_TOK + c8 * 8, KVs + i * 8);
    }
    __builtin_amdgcn_s_wait_asynccnt(0);
#else
    for (int i = tid; i < HPAD * 128 / 8; i += 256) {
      int d = i >> 4, c8 = i & 15;
      ((uint4*)KVs)[i] = *(const uint4*)(vsrc + (long)d * T_TOK + c8 * 8);
    }
#endif
    __syncthreads();

    // O += P @ V   (K dim = 128 keys -> 4 WMMA steps)
#pragma unroll
    for (int kk = 0; kk < 4; ++kk) {
      v16bf ap = ldfrag(Pw, lane15, 128, kk * 32, lane);
#pragma unroll
      for (int d = 0; d < 6; ++d) {
        v16bf bv = ldfrag(KVs, d * 16 + lane15, 128, kk * 32, lane);
        oacc[d] = wmma_bf16(ap, bv, oacc[d]);
      }
    }
  }

  // normalize and write bf16 attention output [T][D]
#pragma unroll
  for (int d = 0; d < 6; ++d) {
    int col = d * 16 + lane15;
    if (col < HDIM) {
#pragma unroll
      for (int i = 0; i < 8; ++i) {
        long row = (long)qb * 128 + w * 16 + i + 8 * kh;
        float o = oacc[d][i] / lrun[i];
        outA[row * D_MODEL + head * HDIM + col] = f2bf(o);
      }
    }
  }
}

// ---------- host launcher ----------
extern "C" void kernel_launch(void* const* d_in, const int* in_sizes, int n_in,
                              void* d_out, int out_size, void* d_ws, size_t ws_size,
                              hipStream_t stream) {
  (void)in_sizes; (void)n_in; (void)out_size; (void)ws_size;
  const float* x     = (const float*)d_in[0];
  const float* rope  = (const float*)d_in[1];
  const float* n1s   = (const float*)d_in[2];
  const float* wqkv  = (const float*)d_in[3];
  const float* bqkv  = (const float*)d_in[4];
  const float* wproj = (const float*)d_in[5];
  const float* bproj = (const float*)d_in[6];
  const float* n2s   = (const float*)d_in[7];
  const float* fc1w  = (const float*)d_in[8];
  const float* fc1b  = (const float*)d_in[9];
  const float* fc2w  = (const float*)d_in[10];
  const float* fc2b  = (const float*)d_in[11];
  const int*   seg   = (const int*)d_in[12];
  float* out = (float*)d_out;

  // workspace layout (bytes, all 256-aligned)
  char* ws = (char*)d_ws;
  unsigned short* wqkv_bf  = (unsigned short*)ws;  ws += (size_t)D_MODEL * 3 * D_MODEL * 2;
  unsigned short* wproj_bf = (unsigned short*)ws;  ws += (size_t)D_MODEL * D_MODEL * 2;
  unsigned short* fc1_bf   = (unsigned short*)ws;  ws += (size_t)D_MODEL * D_FF * 2;
  unsigned short* fc2_bf   = (unsigned short*)ws;  ws += (size_t)D_FF * D_MODEL * 2;
  unsigned short* xn_bf    = (unsigned short*)ws;  ws += (size_t)T_TOK * D_MODEL * 2;  // reused as xn2
  unsigned short* Qb       = (unsigned short*)ws;  ws += (size_t)NHEAD * T_TOK * HPAD * 2;
  unsigned short* Kb       = (unsigned short*)ws;  ws += (size_t)NHEAD * T_TOK * HPAD * 2;
  unsigned short* Vt       = (unsigned short*)ws;  ws += (size_t)NHEAD * HPAD * T_TOK * 2;
  char* R = ws;  // 42,467,328-byte region, reused across phases:
  float*          qkv_f32  = (float*)R;                         // phase 1 (T x 3456 f32)
  unsigned short* attnA    = (unsigned short*)R;                // phase 2 (T x D bf16)
  unsigned short* h_bf     = (unsigned short*)R;                // phase 4 (T x I bf16)
  float*          x1       = (float*)(R + (size_t)28311552);    // residual (T x D f32)

  // 1) weights -> bf16
  cvt_bf16_kernel<<<2048, 256, 0, stream>>>(wqkv,  wqkv_bf,  (long)D_MODEL * 3 * D_MODEL);
  cvt_bf16_kernel<<<2048, 256, 0, stream>>>(wproj, wproj_bf, (long)D_MODEL * D_MODEL);
  cvt_bf16_kernel<<<2048, 256, 0, stream>>>(fc1w,  fc1_bf,   (long)D_MODEL * D_FF);
  cvt_bf16_kernel<<<2048, 256, 0, stream>>>(fc2w,  fc2_bf,   (long)D_FF * D_MODEL);

  // 2) xn = rmsnorm(x) (bf16)
  rmsnorm_kernel<<<T_TOK, 256, 0, stream>>>(x, n1s, xn_bf);

  // 3) qkv = xn @ wqkv + bqkv (f32)
  gemm_bf16_wmma<0><<<dim3(27, 24), 256, 0, stream>>>(
      xn_bf, wqkv_bf, bqkv, nullptr, qkv_f32, nullptr, T_TOK, 3 * D_MODEL, D_MODEL);

  // 4) rope + pack q/k/v (bf16, head-dim padded to 96; sm_scale folded into q)
  rope_pack_kernel<<<T_TOK, 256, 0, stream>>>(qkv_f32, rope, Qb, Kb, Vt);

  // 5) flash attention (writes bf16 attn output over qkv region)
  const int attn_lds = (128 * HPAD + 128 * HPAD + 8 * 16 * 128) * 2 + 256 * 4;
  attn_kernel<<<dim3(T_TOK / 128, NHEAD), 256, attn_lds, stream>>>(
      Qb, Kb, Vt, seg, attnA);

  // 6) x1 = x + attn @ wproj + bproj (f32)
  gemm_bf16_wmma<2><<<dim3(9, 24), 256, 0, stream>>>(
      attnA, wproj_bf, bproj, x, x1, nullptr, T_TOK, D_MODEL, D_MODEL);

  // 7) xn2 = rmsnorm(x1) (bf16, reuses xn buffer)
  rmsnorm_kernel<<<T_TOK, 256, 0, stream>>>(x1, n2s, xn_bf);

  // 8) h = gelu(xn2 @ fc1_w + fc1_b) (bf16, over attnA region)
  gemm_bf16_wmma<1><<<dim3(34, 24), 256, 0, stream>>>(
      xn_bf, fc1_bf, fc1b, nullptr, nullptr, h_bf, T_TOK, D_FF, D_MODEL);

  // 9) out = x1 + h @ fc2_w + fc2_b (f32)
  gemm_bf16_wmma<2><<<dim3(9, 24), 256, 0, stream>>>(
      h_bf, fc2_bf, fc2b, x1, out, nullptr, T_TOK, D_MODEL, D_FF);
}
